// Recurrent_70549132804199
// MI455X (gfx1250) — matrix operand order repaired and checked by reading, not compiled, not run on hardware
//
#include <hip/hip_runtime.h>
#include <hip/hip_bf16.h>
#include <cstdint>
#include <cstddef>

// Problem dims (match reference)
#define B_   32
#define T_   512
#define H_   1024
#define L_   2
#define G4_  (4 * H_)

// GEMM tiling
#define CHUNK  128         // K-chunk staged in LDS (bf16 elements)
#define APAD   8           // LDS row pad (bf16) = 16B = 4 DWORDs (matches TDM pad)
#define NCHUNK (H_ / CHUNK)

typedef __attribute__((ext_vector_type(16))) __bf16       v16bf;
typedef __attribute__((ext_vector_type(8)))  float        v8f;
typedef __attribute__((ext_vector_type(4)))  unsigned int u32x4;
typedef __attribute__((ext_vector_type(8)))  int          i32x8;
typedef __attribute__((ext_vector_type(4)))  int          i32x4;

union BFrag {
    uint4 q[2];
    v16bf v;
};

#if defined(__has_builtin)
#  if __has_builtin(__builtin_amdgcn_tensor_load_to_lds) && \
      __has_builtin(__builtin_amdgcn_s_wait_tensorcnt)
#    define USE_TDM 1
#  endif
#endif
#ifndef USE_TDM
#  define USE_TDM 0
#endif

#if USE_TDM
// ---------------------------------------------------------------------------
// Tensor Data Mover: DMA a rows x cols bf16 2-D tile (row stride in elements)
// from global memory into LDS. Uses D# pad_enable to insert 4 DWORDs (16 B)
// of LDS padding every 64 DWORDs (256 B = one 128-elem bf16 row), matching
// the CHUNK+APAD LDS layout. Wave-level op; issue from one wave only.
// 6-arg builtin (clang-23 / therock-10.0 lane): groups 2,3 are i32x4,
// group 4 (5-D/extra) is i32x8, then cpol.
// ---------------------------------------------------------------------------
__device__ __forceinline__ void tdm_load_tile_bf16(unsigned lds_byte_off,
                                                   const __bf16* gptr,
                                                   unsigned rows, unsigned cols,
                                                   unsigned long long stride_elems)
{
    const unsigned long long ga = (unsigned long long)(uintptr_t)gptr;
    u32x4 g0;
    g0.x = 1u;                                    // count=1, user descriptor
    g0.y = lds_byte_off;                          // LDS dest (bytes)
    g0.z = (unsigned)(ga & 0xFFFFFFFFull);        // global_addr[31:0]
    g0.w = (unsigned)((ga >> 32) & 0x1FFFFFFull)  // global_addr[56:32]
         | (2u << 30);                            // type = 2 ("image")
    i32x8 g1;
    // wg_mask=0 | data_size=1 (2B) | pad_enable | pad_interval=5 (64 DW)
    // | pad_amount=3 (4 DW)
    g1[0] = (int)((1u << 16) | (1u << 20) | (5u << 22) | (3u << 25));
    g1[1] = (int)((cols & 0xFFFFu) << 16);                          // tensor_dim0 lo
    g1[2] = (int)(((cols >> 16) & 0xFFFFu) | ((rows & 0xFFFFu) << 16)); // d0 hi | d1 lo
    g1[3] = (int)(((rows >> 16) & 0xFFFFu) | ((cols & 0xFFFFu) << 16)); // d1 hi | tile_dim0
    g1[4] = (int)(rows & 0xFFFFu);                                  // tile_dim1 | tile_dim2=0
    g1[5] = (int)(stride_elems & 0xFFFFFFFFull);                    // dim0_stride lo
    g1[6] = (int)((stride_elems >> 32) & 0xFFFFull);                // dim0_stride hi
    g1[7] = 0;
    const i32x4 gz4 = {0, 0, 0, 0};               // groups 2/3 unused (2-D tensor)
    const i32x8 gz8 = {0, 0, 0, 0, 0, 0, 0, 0};   // extra group unused
    __builtin_amdgcn_tensor_load_to_lds(g0, g1, gz4, gz4, gz8, 0);
}
#endif

// ---------------------------------------------------------------------------
// One-time (per call) conversions: weights/input -> bf16, bias fusion,
// h0 -> bf16 ping-pong buffer 0. All bf16 streams stay L2-resident (192 MB).
// ---------------------------------------------------------------------------
__global__ void lstm_cvt(const float* __restrict__ Wih, const float* __restrict__ Whh,
                         const float* __restrict__ bih, const float* __restrict__ bhh,
                         const float* __restrict__ inputs, const float* __restrict__ h0,
                         __bf16* __restrict__ wih_bf, __bf16* __restrict__ whh_bf,
                         float* __restrict__ bias, __bf16* __restrict__ xin_bf,
                         __bf16* __restrict__ hbf)
{
    const long t0 = (long)blockIdx.x * blockDim.x + threadIdx.x;
    const long gs = (long)gridDim.x * blockDim.x;

    const long NW = (long)L_ * G4_ * H_;
    for (long i = t0; i < NW; i += gs) {
        wih_bf[i] = (__bf16)Wih[i];
        whh_bf[i] = (__bf16)Whh[i];
    }
    const long NX = (long)B_ * T_ * H_;
    for (long i = t0; i < NX; i += gs) xin_bf[i] = (__bf16)inputs[i];

    const long NH = (long)L_ * B_ * H_;
    for (long i = t0; i < NH; i += gs) {
        const long l = i / ((long)B_ * H_);
        const long r = i % ((long)B_ * H_);
        hbf[(l * 2 + 0) * (long)B_ * H_ + r] = (__bf16)h0[i];   // ping-pong buf 0
    }
    if (t0 < L_ * G4_) bias[t0] = bih[t0] + bhh[t0];
}

// ---------------------------------------------------------------------------
// One fused LSTM layer-step. Block = 256 thr = 8 waves; block covers all
// M=32 rows x 16 hidden cols x 4 gates (one 16x16 gate-tile per wave).
// Grid = H/16 = 64 blocks -> 512 waves per step (max tile parallelism).
// A tiles (x and h_prev, both bf16) are TDM-DMA'd into double-buffered LDS;
// weight B fragments stream straight from L2 into WMMA.
// ---------------------------------------------------------------------------
__global__ void __launch_bounds__(256)
lstm_step(const __bf16* __restrict__ xbf, long x_stride,
          const __bf16* __restrict__ Wih,       // [4H, H] bf16 layer slice
          const __bf16* __restrict__ Whh,       // [4H, H] bf16 layer slice
          const float*  __restrict__ bias,      // [4H] fused bias, layer slice
          const __bf16* __restrict__ hbf_prev,  // [B, H] bf16
          __bf16* __restrict__ hbf_next,        // [B, H] bf16 (ping-pong)
          float*  __restrict__ h_fp32,          // [B, H] fp32 shadow (for h_T)
          float*  __restrict__ c_state,         // [B, H] fp32, in place
          float*  __restrict__ out)             // nullable: d_out + t*H, stride T*H
{
    __shared__ __align__(16) __bf16 Abuf[2][2][B_][CHUNK + APAD]; // [buf][x|h]
    __shared__ float G[4][B_][17];                                // gate exchange

    const int tid    = threadIdx.x;
    const int lane   = tid & 31;
    const int wid    = tid >> 5;
    const int g      = wid & 3;        // gate i/f/g/o
    const int mi     = wid >> 2;       // M tile (rows 0..15 / 16..31)
    const int hc0    = blockIdx.x * 16;
    const int lh     = lane & 15;
    const int hiHalf = lane >> 4;

    const int col = g * H_ + hc0 + lh; // weight row / gate column for this lane
    v8f acc = {};

    auto compute = [&](int cb, int kbase) {
        if (kbase + CHUNK < H_) {
            __builtin_prefetch(Wih + (long)col * H_ + kbase + CHUNK, 0, 1);
            __builtin_prefetch(Whh + (long)col * H_ + kbase + CHUNK, 0, 1);
        }
        #pragma unroll
        for (int ks = 0; ks < CHUNK; ks += 32) {
            #pragma unroll
            for (int s = 0; s < 2; ++s) {
                __bf16 (*A)[CHUNK + APAD] = Abuf[cb][s];
                const __bf16* Wb          = s ? Whh : Wih;

                // A frag: 16-bit A 16x32 layout (two ds_load_b128 per lane)
                BFrag a;
                const int arow = mi * 16 + lh;
                const int ak   = ks + hiHalf * 8;
                a.q[0] = *(const uint4*)&A[arow][ak];
                a.q[1] = *(const uint4*)&A[arow][ak + 16];

                // B frag: K = 8h..8h+7 and 16+8h..16+8h+7 per lane (two global b128)
                const __bf16* wp = Wb + (long)col * H_ + kbase + ks + hiHalf * 8;
                BFrag b;
                b.q[0] = *(const uint4*)&wp[0];
                b.q[1] = *(const uint4*)&wp[16];

                acc = __builtin_amdgcn_wmma_f32_16x16x32_bf16(
                    false, a.v, false, b.v, (short)0, acc, false, false);
            }
        }
    };

#if USE_TDM
    auto stage = [&](int kc, int cb) {
        tdm_load_tile_bf16((unsigned)(uintptr_t)&Abuf[cb][0][0][0],
                           xbf + kc, B_, CHUNK, (unsigned long long)x_stride);
        tdm_load_tile_bf16((unsigned)(uintptr_t)&Abuf[cb][1][0][0],
                           hbf_prev + kc, B_, CHUNK, (unsigned long long)H_);
    };
    if (wid == 0) stage(0, 0);
    for (int ic = 0; ic < NCHUNK; ++ic) {
        const int cb = ic & 1;
        if (wid == 0) __builtin_amdgcn_s_wait_tensorcnt(0);  // TDM(ic) complete
        __syncthreads();            // data visible; prior readers of cb^1 done
        if (wid == 0 && ic + 1 < NCHUNK) stage((ic + 1) * CHUNK, cb ^ 1);
        compute(cb, ic * CHUNK);    // overlaps with TDM(ic+1)
    }
    __syncthreads();
#else
    // Fallback: cooperative bf16 copy (two b128 per source per thread)
    for (int ic = 0; ic < NCHUNK; ++ic) {
        const int kc  = ic * CHUNK;
        const int row = tid >> 3;
        const int seg = (tid & 7) * 16;
        const __bf16* xs = xbf      + (long)row * x_stride + kc + seg;
        const __bf16* hs = hbf_prev + (long)row * H_       + kc + seg;
        *(uint4*)&Abuf[0][0][row][seg]     = *(const uint4*)xs;
        *(uint4*)&Abuf[0][0][row][seg + 8] = *(const uint4*)(xs + 8);
        *(uint4*)&Abuf[0][1][row][seg]     = *(const uint4*)hs;
        *(uint4*)&Abuf[0][1][row][seg + 8] = *(const uint4*)(hs + 8);
        __syncthreads();
        compute(0, kc);
        __syncthreads();
    }
#endif

    // ---- gate exchange through LDS, then fused cell update ----
    #pragma unroll
    for (int r = 0; r < 8; ++r) {
        const int row = mi * 16 + r + hiHalf * 8;   // C/D layout: VGPR r -> M=r/r+8
        G[g][row][lh] = acc[r];
    }
    __syncthreads();

    #pragma unroll
    for (int e = 0; e < 2; ++e) {
        const int idx = tid + e * 256;              // 0..511 = 32 rows x 16 cols
        const int row = idx >> 4;
        const int cl  = idx & 15;
        const int hcol = hc0 + cl;
        float gi = G[0][row][cl] + bias[0 * H_ + hcol];
        float gf = G[1][row][cl] + bias[1 * H_ + hcol];
        float gg = G[2][row][cl] + bias[2 * H_ + hcol];
        float go = G[3][row][cl] + bias[3 * H_ + hcol];
        gi = 1.0f / (1.0f + __expf(-gi));
        gf = 1.0f / (1.0f + __expf(-gf));
        gg = tanhf(gg);
        go = 1.0f / (1.0f + __expf(-go));

        const long off  = (long)row * H_ + hcol;
        const float c_n = gf * c_state[off] + gi * gg;
        const float h_n = go * tanhf(c_n);
        c_state[off]  = c_n;
        hbf_next[off] = (__bf16)h_n;
        h_fp32[off]   = h_n;
        if (out) out[(long)row * (T_ * H_) + hcol] = h_n;   // outputs[b, t, h]
    }
}

// ---------------------------------------------------------------------------
// Host orchestration (graph-capture friendly: only kernels + async d2d).
// ---------------------------------------------------------------------------
extern "C" void kernel_launch(void* const* d_in, const int* in_sizes, int n_in,
                              void* d_out, int out_size, void* d_ws, size_t ws_size,
                              hipStream_t stream)
{
    const float* inputs = (const float*)d_in[0];   // [B, T, H]
    const float* h0     = (const float*)d_in[1];   // [L, B, H]
    const float* c0     = (const float*)d_in[2];   // [L, B, H]
    const float* Wih    = (const float*)d_in[3];   // [L, 4H, H]
    const float* Whh    = (const float*)d_in[4];   // [L, 4H, H]
    const float* bih    = (const float*)d_in[5];   // [L, 4H]
    const float* bhh    = (const float*)d_in[6];   // [L, 4H]
    float* out = (float*)d_out;                    // outputs | h_T | c_T
    (void)in_sizes; (void)n_in; (void)out_size; (void)ws_size;

    // workspace carve-up (~68 MB)
    char* ws = (char*)d_ws;
    size_t off = 0;
    __bf16* wih_bf = (__bf16*)(ws + off); off += (size_t)L_ * G4_ * H_ * sizeof(__bf16);
    __bf16* whh_bf = (__bf16*)(ws + off); off += (size_t)L_ * G4_ * H_ * sizeof(__bf16);
    __bf16* xin_bf = (__bf16*)(ws + off); off += (size_t)B_ * T_ * H_ * sizeof(__bf16);
    float*  bias   = (float*) (ws + off); off += (size_t)L_ * G4_ * sizeof(float);
    __bf16* hbf    = (__bf16*)(ws + off); off += (size_t)L_ * 2 * B_ * H_ * sizeof(__bf16);
    float*  hfp    = (float*) (ws + off); off += (size_t)L_ * B_ * H_ * sizeof(float);
    float*  cbuf   = (float*) (ws + off); off += (size_t)L_ * B_ * H_ * sizeof(float);

    lstm_cvt<<<8192, 256, 0, stream>>>(Wih, Whh, bih, bhh, inputs, h0,
                                       wih_bf, whh_bf, bias, xin_bf, hbf);

    const size_t stateSz = (size_t)B_ * H_ * sizeof(float);
    for (int l = 0; l < L_; ++l) {
        (void)hipMemcpyAsync(cbuf + (size_t)l * B_ * H_, c0 + (size_t)l * B_ * H_,
                             stateSz, hipMemcpyDeviceToDevice, stream);
    }

    const dim3 grid(H_ / 16), block(256);
    const size_t bh = (size_t)B_ * H_;
    for (int t = 0; t < T_; ++t) {
        const int cur = t & 1;
        // layer 0: x from pre-converted bf16 inputs[:, t, :] (row stride T*H)
        const __bf16* h0p = hbf + (0 * 2 + cur)       * bh;
        __bf16*       h0n = hbf + (0 * 2 + (cur ^ 1)) * bh;
        lstm_step<<<grid, block, 0, stream>>>(
            xin_bf + (size_t)t * H_, (long)T_ * H_,
            wih_bf, whh_bf, bias,
            h0p, h0n, hfp, cbuf, nullptr);

        // layer 1: x = layer-0's fresh bf16 h; writes outputs[:, t, :]
        const __bf16* h1p = hbf + (1 * 2 + cur)       * bh;
        __bf16*       h1n = hbf + (1 * 2 + (cur ^ 1)) * bh;
        lstm_step<<<grid, block, 0, stream>>>(
            h0n, (long)H_,
            wih_bf + (size_t)G4_ * H_, whh_bf + (size_t)G4_ * H_, bias + G4_,
            h1p, h1n, hfp + bh, cbuf + bh, out + (size_t)t * H_);
    }

    // final states: h_T from fp32 shadow (written every step), c_T in place
    float* outH = out + (size_t)B_ * T_ * H_;
    float* outC = outH + (size_t)L_ * B_ * H_;
    for (int l = 0; l < L_; ++l) {
        (void)hipMemcpyAsync(outH + (size_t)l * bh, hfp  + (size_t)l * bh,
                             stateSz, hipMemcpyDeviceToDevice, stream);
        (void)hipMemcpyAsync(outC + (size_t)l * bh, cbuf + (size_t)l * bh,
                             stateSz, hipMemcpyDeviceToDevice, stream);
    }
}
